// ConvViT_61881888801215
// MI455X (gfx1250) — compile-verified, matrix operand order checked by reading, not run on hardware
//
#include <hip/hip_runtime.h>
#include <hip/hip_bf16.h>
#include <math.h>

typedef __bf16 bf16_t;
typedef bf16_t v16bf_t __attribute__((ext_vector_type(16)));
typedef bf16_t v8bf_t  __attribute__((ext_vector_type(8)));
typedef float  v8f_t   __attribute__((ext_vector_type(8)));

#define DEV __device__ __forceinline__

constexpr int B_   = 4;
constexpr int N_   = 1280;
constexpr int NM_  = 256;
constexpr int NS_  = 1024;
constexpr int DIM_ = 768;
constexpr int H_   = 12;
constexpr int D_   = 64;
constexpr int MEM_ = 1280;
constexpr int KF_  = 2560;     // MEM + N
constexpr int R_   = 8;
constexpr int TOPM_ = 128;     // 50% of 256
constexpr int TOPS_ = 640;     // 25% of 2560
constexpr float LSC_ = 0.125f; // 1/R
constexpr float QSC_ = 0.125f; // D^-0.5

// ---------------- helpers ----------------

DEV bf16_t f2bf(float f) {
  unsigned u = __float_as_uint(f);
  unsigned r = u + 0x7FFFu + ((u >> 16) & 1u);   // RNE
  unsigned short h = (unsigned short)(r >> 16);
  bf16_t b;
  __builtin_memcpy(&b, &h, sizeof(h));
  return b;
}

DEV unsigned fkey(float f) {   // order-preserving float->uint
  unsigned u = __float_as_uint(f);
  return (u & 0x80000000u) ? ~u : (u | 0x80000000u);
}

DEV int   wred_i(int v)   { for (int o = 16; o; o >>= 1) v += __shfl_xor(v, o, 32); return v; }
DEV float wred_f(float v) { for (int o = 16; o; o >>= 1) v += __shfl_xor(v, o, 32); return v; }
DEV float wred_m(float v) { for (int o = 16; o; o >>= 1) v = fmaxf(v, __shfl_xor(v, o, 32)); return v; }

// A/B fragment for v_wmma_f32_16x16x32_bf16 from a row-major source row:
// lane<16: K 0..7 and 16..23 ; lane>=16: K 8..15 and 24..31 (caller adds koff)
DEV v16bf_t load_frag_g(const bf16_t* p) {
  v8bf_t lo = *(const v8bf_t*)p;
  v8bf_t hi = *(const v8bf_t*)(p + 16);
  v16bf_t r;
#pragma unroll
  for (int i = 0; i < 8; i++) { r[i] = lo[i]; r[8 + i] = hi[i]; }
  return r;
}

DEV v16bf_t frag_from_f32(const float* p) {
  v16bf_t r;
#pragma unroll
  for (int i = 0; i < 8; i++) { r[i] = f2bf(p[i]); r[8 + i] = f2bf(p[16 + i]); }
  return r;
}

DEV v8f_t wmma_bf16(v16bf_t a, v16bf_t b, v8f_t c) {
  return __builtin_amdgcn_wmma_f32_16x16x32_bf16(false, a, false, b, (short)0, c, false, false);
}

// ---------------- prep kernels ----------------

__global__ void cvt_bf16_kernel(const float* __restrict__ in, bf16_t* __restrict__ out, size_t n) {
  size_t i = (size_t)blockIdx.x * blockDim.x + threadIdx.x;
  if (i < n) out[i] = f2bf(in[i]);
}

// W_eff[o][k] = W[o][k] + (1/R) * sum_r Bm[o][r] * A[g(o)][r][k]   (bf16 out)
__global__ void fold_w_kernel(const float* __restrict__ W, const float* __restrict__ A,
                              const float* __restrict__ Bm, bf16_t* __restrict__ out,
                              int NR, int K, int GS) {
  size_t idx = (size_t)blockIdx.x * blockDim.x + threadIdx.x;
  if (idx >= (size_t)NR * K) return;
  int o = (int)(idx / K), k = (int)(idx % K);
  int g = o / GS;
  float acc = W[idx];
  const float* br = Bm + (size_t)o * R_;
  const float* ar = A + (size_t)g * R_ * K;
#pragma unroll
  for (int r = 0; r < R_; r++) acc = fmaf(LSC_ * br[r], ar[r * K + k], acc);
  out[idx] = f2bf(acc);
}

__global__ void fold_idk_kernel(const float* __restrict__ W, const float* __restrict__ A,
                                const float* __restrict__ Bm, float* __restrict__ out) {
  int idx = blockIdx.x * blockDim.x + threadIdx.x;
  if (idx >= H_ * DIM_) return;
  int h = idx / DIM_, k = idx % DIM_;
  float acc = W[idx];
#pragma unroll
  for (int r = 0; r < R_; r++) acc = fmaf(LSC_ * Bm[h * R_ + r], A[r * DIM_ + k], acc);
  out[idx] = acc;
}

__global__ void bcat_kernel(const float* __restrict__ pB, float* __restrict__ Bcat) {
  int idx = blockIdx.x * blockDim.x + threadIdx.x;
  if (idx >= DIM_ * 32) return;
  int j = idx % 32, c = idx / 32;
  int e = j >> 3, r = j & 7;
  Bcat[idx] = pB[((size_t)e * DIM_ + c) * R_ + r];
}

__global__ void scatter_mem_kernel(const float* __restrict__ mk, const float* __restrict__ mv,
                                   bf16_t* __restrict__ kfull, bf16_t* __restrict__ vT) {
  size_t idx = (size_t)blockIdx.x * blockDim.x + threadIdx.x;
  size_t total = (size_t)B_ * H_ * MEM_ * D_;
  if (idx >= total) return;
  int d = (int)(idx % D_);
  int m = (int)((idx / D_) % MEM_);
  size_t bh = idx / ((size_t)D_ * MEM_);
  kfull[(bh * KF_ + m) * D_ + d] = f2bf(mk[idx]);
  vT[(bh * D_ + d) * (size_t)KF_ + m] = f2bf(mv[idx]);
}

// ---------------- generic GEMM: C[M,N] = A[M,K] @ W[N,K]^T + bias ----------------
// one wave computes a 16x64 tile (1x4 register blocking: A-fragment reused 4x),
// block = (32,4) -> workgroup covers 16 x 256 of C.

__global__ __launch_bounds__(128) void gemm_tt_bf16(
    const bf16_t* __restrict__ A, const bf16_t* __restrict__ W,
    const float* __restrict__ bias, float* __restrict__ C, int M, int N, int K) {
  const int lane = threadIdx.x;
  const int tn0 = (blockIdx.x * blockDim.y + threadIdx.y) * 4;   // first of 4 n-tiles
  const int tm = blockIdx.y;
  if (tn0 * 16 >= N || tm * 16 >= M) return;
  const int row  = lane & 15;
  const int koff = (lane & 16) ? 8 : 0;
  const bf16_t* arow = A + (size_t)(tm * 16 + row) * K + koff;
  const bf16_t* w0   = W + (size_t)(tn0 * 16 + row) * K + koff;
  const size_t ws = (size_t)16 * K;   // stride between n-tiles of W
  v8f_t acc0 = {}, acc1 = {}, acc2 = {}, acc3 = {};
  for (int k0 = 0; k0 < K; k0 += 32) {
    __builtin_prefetch(arow + k0 + 512, 0, 0);       // global_prefetch_b8
    __builtin_prefetch(w0 + k0 + 512, 0, 0);
    v16bf_t af = load_frag_g(arow + k0);
    acc0 = wmma_bf16(af, load_frag_g(w0 + k0), acc0);
    acc1 = wmma_bf16(af, load_frag_g(w0 + ws + k0), acc1);
    acc2 = wmma_bf16(af, load_frag_g(w0 + 2 * ws + k0), acc2);
    acc3 = wmma_bf16(af, load_frag_g(w0 + 3 * ws + k0), acc3);
  }
  const int m0 = tm * 16 + ((lane & 16) ? 8 : 0);
  v8f_t accs[4] = {acc0, acc1, acc2, acc3};
#pragma unroll
  for (int t = 0; t < 4; t++) {
    const int n = (tn0 + t) * 16 + row;
    const float bsv = bias ? bias[n] : 0.0f;
#pragma unroll
    for (int j = 0; j < 8; j++)
      C[(size_t)(m0 + j) * N + n] = accs[t][j] + bsv;
  }
}

// ---------------- qkv scatter ----------------

__global__ void scatter_qkv_kernel(const float* __restrict__ qkv,
                                   bf16_t* __restrict__ qb, bf16_t* __restrict__ kfull,
                                   bf16_t* __restrict__ vT, float* __restrict__ k_m,
                                   float* __restrict__ v_m) {
  size_t idx = (size_t)blockIdx.x * blockDim.x + threadIdx.x;
  size_t total = (size_t)B_ * N_ * 3 * DIM_;
  if (idx >= total) return;
  int o = (int)(idx % (3 * DIM_));
  size_t t = idx / (3 * DIM_);
  int n = (int)(t % N_);
  int b = (int)(t / N_);
  float v = qkv[idx];
  int which = o / DIM_;
  int c = o % DIM_;
  int h = c / D_, d = c % D_;
  size_t bh = (size_t)b * H_ + h;
  if (which == 0) {
    qb[(bh * N_ + n) * D_ + d] = f2bf(v * QSC_);
  } else if (which == 1) {
    kfull[(bh * KF_ + MEM_ + n) * D_ + d] = f2bf(v);
    if (n < NM_) k_m[(bh * NM_ + n) * D_ + d] = v;
  } else {
    if (n < NM_) v_m[(bh * NM_ + n) * D_ + d] = v;
    else vT[(bh * D_ + d) * (size_t)KF_ + MEM_ + n] = f2bf(v);
  }
}

// ---------------- identity-token modulation ----------------

__global__ void idk_scale_kernel(const float* __restrict__ idt, const float* __restrict__ Wk,
                                 const float* __restrict__ bk, float* __restrict__ scale_m) {
  int idx = blockIdx.x * blockDim.x + threadIdx.x;
  if (idx >= B_ * NM_ * H_) return;
  int h = idx % H_;
  int n = (idx / H_) % NM_;
  int b = idx / (H_ * NM_);
  const float* xr = idt + ((size_t)b * NM_ + n) * DIM_;
  const float* wr = Wk + (size_t)h * DIM_;
  float acc = bk[h];
  for (int k = 0; k < DIM_; k++) acc = fmaf(xr[k], wr[k], acc);
  scale_m[idx] = 1.0f + tanhf(acc);
}

__global__ void combine_mid_kernel(const float* __restrict__ k_m, const float* __restrict__ v_m,
                                   const float* __restrict__ scale_m, const float* __restrict__ idv,
                                   float* __restrict__ out_kmid, float* __restrict__ out_vmid,
                                   bf16_t* __restrict__ kmid_b, bf16_t* __restrict__ vmidT_b,
                                   bf16_t* __restrict__ vT) {
  size_t idx = (size_t)blockIdx.x * blockDim.x + threadIdx.x;
  size_t total = (size_t)B_ * H_ * NM_ * D_;
  if (idx >= total) return;
  int d = (int)(idx % D_);
  int n = (int)((idx / D_) % NM_);
  int h = (int)((idx / ((size_t)D_ * NM_)) % H_);
  int b = (int)(idx / ((size_t)D_ * NM_ * H_));
  float sc = scale_m[((size_t)b * NM_ + n) * H_ + h];
  float kv = k_m[idx] * sc;
  out_kmid[idx] = kv;
  kmid_b[idx] = f2bf(kv);
  float vv = v_m[idx] + idv[((size_t)b * NM_ + n) * DIM_ + h * D_ + d];
  out_vmid[idx] = vv;
  size_t bh = (size_t)b * H_ + h;
  vmidT_b[(bh * D_ + d) * (size_t)NM_ + n] = f2bf(vv);
  vT[(bh * D_ + d) * (size_t)KF_ + MEM_ + n] = f2bf(vv);
}

// ---------------- attention over memory tokens (256 q x 256 k per b,h) ----------------
// grid = B*H ; block = (32,8) ; dynamic LDS = 8 * 16*256 * 4 = 128KB

__global__ __launch_bounds__(256) void attn_m_kernel(
    const bf16_t* __restrict__ qb, const bf16_t* __restrict__ kmid,
    const bf16_t* __restrict__ vmidT, float* __restrict__ xo) {
  extern __shared__ float smem[];
  const int lane = threadIdx.x, wave = threadIdx.y;
  const int bh = blockIdx.x;
  const int b = bh / H_, h = bh % H_;
  const bf16_t* Q  = qb    + (size_t)bh * N_  * D_;
  const bf16_t* Kp = kmid  + (size_t)bh * NM_ * D_;
  const bf16_t* Vt = vmidT + (size_t)bh * D_  * NM_;
  float* Sw = smem + (size_t)wave * 16 * NM_;
  const int row  = lane & 15;
  const int koff = (lane & 16) ? 8 : 0;
  const int mb   = (lane & 16) ? 8 : 0;

  for (int qt = wave; qt < NM_ / 16; qt += 8) {
    // S = Q_tile @ Kmid^T  (16 x 256)
    for (int kt = 0; kt < NM_ / 16; ++kt) {
      v8f_t acc = {};
#pragma unroll
      for (int kk = 0; kk < D_; kk += 32)
        acc = wmma_bf16(load_frag_g(Q  + (size_t)(qt * 16 + row) * D_ + kk + koff),
                        load_frag_g(Kp + (size_t)(kt * 16 + row) * D_ + kk + koff), acc);
#pragma unroll
      for (int j = 0; j < 8; j++) Sw[(mb + j) * NM_ + kt * 16 + row] = acc[j];
    }
    // exact top-128 softmax, one row per lane (lanes 0..15)
    if (lane < 16) {
      float* Srow = Sw + lane * NM_;
      unsigned lo = 0u, hi = 0xFFFFFFFFu;
      while (lo < hi) {
        unsigned mid = (unsigned)(((unsigned long long)lo + hi + 1ull) >> 1);
        int c = 0;
        for (int i = 0; i < NM_; ++i) c += (fkey(Srow[i]) >= mid) ? 1 : 0;
        if (c >= TOPM_) lo = mid; else hi = mid - 1;
      }
      float mx = -3.4e38f;
      for (int i = 0; i < NM_; ++i) mx = fmaxf(mx, Srow[i]);
      float sm = 0.0f;
      for (int i = 0; i < NM_; ++i)
        if (fkey(Srow[i]) >= lo) sm += expf(Srow[i] - mx);
      float inv = 1.0f / sm;
      for (int i = 0; i < NM_; ++i)
        Srow[i] = (fkey(Srow[i]) >= lo) ? expf(Srow[i] - mx) * inv : 0.0f;
    }
    // x_m tile = P @ Vmid  (16 x 64), Vt rows are d (contiguous over keys)
    for (int nt = 0; nt < 4; ++nt) {
      v8f_t acc = {};
      for (int k0 = 0; k0 < NM_; k0 += 32)
        acc = wmma_bf16(frag_from_f32(Sw + row * NM_ + k0 + koff),
                        load_frag_g(Vt + (size_t)(nt * 16 + row) * NM_ + k0 + koff), acc);
      const int q = qt * 16 + mb;
      const int d = nt * 16 + row;
#pragma unroll
      for (int j = 0; j < 8; j++)
        xo[((size_t)b * N_ + q + j) * DIM_ + h * D_ + d] = acc[j];
    }
  }
}

// ---------------- attention over sequence tokens (1024 q x 2560 k per b,h) ----------------
// grid = (NS/16, H, B) ; block = (32,8) ; dynamic LDS = 16 * 2560 * 4 = 160KB

__global__ __launch_bounds__(256) void attn_s_kernel(
    const bf16_t* __restrict__ qb, const bf16_t* __restrict__ kfull,
    const bf16_t* __restrict__ vT, float* __restrict__ xo) {
  extern __shared__ float smem[];
  float* S = smem;
  const int lane = threadIdx.x, wave = threadIdx.y;
  const int qt = blockIdx.x;
  const int h = blockIdx.y, b = blockIdx.z;
  const size_t bh = (size_t)b * H_ + h;
  const bf16_t* Q  = qb + (bh * N_ + NM_ + (size_t)qt * 16) * D_;
  const bf16_t* Kf = kfull + bh * KF_ * D_;
  const bf16_t* Vt = vT + bh * D_ * KF_;
  const int row  = lane & 15;
  const int koff = (lane & 16) ? 8 : 0;
  const int mb   = (lane & 16) ? 8 : 0;

  // scores: each wave covers key tiles wave, wave+8, ... (Q fragments are loop-invariant)
  for (int kt = wave; kt < KF_ / 16; kt += 8) {
    v8f_t acc = {};
#pragma unroll
    for (int kk = 0; kk < D_; kk += 32)
      acc = wmma_bf16(load_frag_g(Q  + (size_t)row * D_ + kk + koff),
                      load_frag_g(Kf + (size_t)(kt * 16 + row) * D_ + kk + koff), acc);
#pragma unroll
    for (int j = 0; j < 8; j++) S[(size_t)(mb + j) * KF_ + kt * 16 + row] = acc[j];
  }
  __syncthreads();

  // exact top-640 softmax: wave w owns rows 2w, 2w+1; 32 lanes cooperate per row
  for (int r = wave * 2; r < wave * 2 + 2; ++r) {
    float* Srow = S + (size_t)r * KF_;
    unsigned lo = 0u, hi = 0xFFFFFFFFu;
    for (int it = 0; it < 33 && lo < hi; ++it) {
      unsigned mid = (unsigned)(((unsigned long long)lo + hi + 1ull) >> 1);
      int c = 0;
      for (int i = lane; i < KF_; i += 32) c += (fkey(Srow[i]) >= mid) ? 1 : 0;
      c = wred_i(c);
      if (c >= TOPS_) lo = mid; else hi = mid - 1;
    }
    float mx = -3.4e38f;
    for (int i = lane; i < KF_; i += 32) mx = fmaxf(mx, Srow[i]);
    mx = wred_m(mx);
    float sm = 0.0f;
    for (int i = lane; i < KF_; i += 32)
      if (fkey(Srow[i]) >= lo) sm += expf(Srow[i] - mx);
    sm = wred_f(sm);
    float inv = 1.0f / sm;
    for (int i = lane; i < KF_; i += 32)
      Srow[i] = (fkey(Srow[i]) >= lo) ? expf(Srow[i] - mx) * inv : 0.0f;
  }
  __syncthreads();

  // x_s tile = P @ Vfull  (16 x 64): waves 0..3, one 16-wide d-tile each
  if (wave < 4) {
    v8f_t acc = {};
    for (int k0 = 0; k0 < KF_; k0 += 32)
      acc = wmma_bf16(frag_from_f32(S + (size_t)row * KF_ + k0 + koff),
                      load_frag_g(Vt + (size_t)(wave * 16 + row) * KF_ + k0 + koff), acc);
    const int q = NM_ + qt * 16 + mb;
    const int d = wave * 16 + row;
#pragma unroll
    for (int j = 0; j < 8; j++)
      xo[((size_t)b * N_ + q + j) * DIM_ + h * D_ + d] = acc[j];
  }
}

// ---------------- routing + routed-LoRA epilogue ----------------

__global__ void route_kernel(const float* __restrict__ xo, const float* __restrict__ rW,
                             float* __restrict__ route) {
  int t = blockIdx.x * blockDim.x + threadIdx.x;
  if (t >= B_ * N_) return;
  const float* xr = xo + (size_t)t * DIM_;
  float lg[4];
#pragma unroll
  for (int e = 0; e < 4; e++) {
    float a = 0.0f;
    const float* wr = rW + (size_t)e * DIM_;
    for (int k = 0; k < DIM_; k++) a = fmaf(xr[k], wr[k], a);
    lg[e] = a;
  }
  float mx = fmaxf(fmaxf(lg[0], lg[1]), fmaxf(lg[2], lg[3]));
  float s = 0.0f;
#pragma unroll
  for (int e = 0; e < 4; e++) { lg[e] = expf(lg[e] - mx); s += lg[e]; }
  float inv = 1.0f / s;
#pragma unroll
  for (int e = 0; e < 4; e++) route[(size_t)t * 4 + e] = lg[e] * inv;
}

__global__ void coef_kernel(const float* __restrict__ xo, const float* __restrict__ pA,
                            const float* __restrict__ route, float* __restrict__ coef) {
  size_t idx = (size_t)blockIdx.x * blockDim.x + threadIdx.x;
  if (idx >= (size_t)B_ * N_ * 32) return;
  int j = (int)(idx % 32);
  size_t t = idx / 32;
  int e = j >> 3;
  const float* xr = xo + t * DIM_;
  const float* ar = pA + (size_t)j * DIM_;
  float a = 0.0f;
  for (int k = 0; k < DIM_; k++) a = fmaf(xr[k], ar[k], a);
  coef[idx] = a * route[t * 4 + e] * LSC_;
}

__global__ void epilogue_kernel(const float* __restrict__ coef, const float* __restrict__ Bcat,
                                float* __restrict__ out) {
  size_t idx = (size_t)blockIdx.x * blockDim.x + threadIdx.x;
  if (idx >= (size_t)B_ * N_ * DIM_) return;
  int c = (int)(idx % DIM_);
  size_t t = idx / DIM_;
  float a = out[idx];
  const float* cf = coef + t * 32;
  const float* bb = Bcat + (size_t)c * 32;
#pragma unroll
  for (int j = 0; j < 32; j++) a = fmaf(cf[j], bb[j], a);
  out[idx] = a;
}

// ---------------- launch ----------------

static inline dim3 g1(size_t n) { return dim3((unsigned)((n + 255) / 256)); }

extern "C" void kernel_launch(void* const* d_in, const int* in_sizes, int n_in,
                              void* d_out, int out_size, void* d_ws, size_t ws_size,
                              hipStream_t stream) {
  (void)in_sizes; (void)n_in; (void)out_size; (void)ws_size;
  const float* x       = (const float*)d_in[0];
  const float* id_tot  = (const float*)d_in[1];
  const float* mem_k   = (const float*)d_in[2];
  const float* mem_v   = (const float*)d_in[3];
  const float* W_qkv   = (const float*)d_in[4];
  const float* b_qkv   = (const float*)d_in[5];
  const float* qkv_A   = (const float*)d_in[6];
  const float* qkv_B   = (const float*)d_in[7];
  const float* W_proj  = (const float*)d_in[8];
  const float* b_proj  = (const float*)d_in[9];
  const float* route_W = (const float*)d_in[10];
  const float* proj_A  = (const float*)d_in[11];
  const float* proj_B  = (const float*)d_in[12];
  const float* W_idk   = (const float*)d_in[13];
  const float* b_idk   = (const float*)d_in[14];
  const float* idk_A   = (const float*)d_in[15];
  const float* idk_B   = (const float*)d_in[16];
  const float* W_idv   = (const float*)d_in[17];
  const float* b_idv   = (const float*)d_in[18];
  const float* idv_A   = (const float*)d_in[19];
  const float* idv_B   = (const float*)d_in[20];

  float* out      = (float*)d_out;
  float* out_kmid = out + (size_t)B_ * N_ * DIM_;
  float* out_vmid = out_kmid + (size_t)B_ * H_ * NM_ * D_;

  char* base = (char*)d_ws;
  size_t off = 0;
  auto alloc = [&](size_t bytes) -> void* {
    off = (off + 255) & ~(size_t)255;
    void* p = base + off;
    off += bytes;
    return p;
  };
  bf16_t* xb      = (bf16_t*)alloc((size_t)B_ * N_ * DIM_ * 2);
  bf16_t* idb     = (bf16_t*)alloc((size_t)B_ * NM_ * DIM_ * 2);
  bf16_t* Wqkv_e  = (bf16_t*)alloc((size_t)3 * DIM_ * DIM_ * 2);
  bf16_t* Widv_e  = (bf16_t*)alloc((size_t)DIM_ * DIM_ * 2);
  float*  Widk_e  = (float*)alloc((size_t)H_ * DIM_ * 4);
  bf16_t* Wproj_b = (bf16_t*)alloc((size_t)DIM_ * DIM_ * 2);
  float*  Bcat    = (float*)alloc((size_t)DIM_ * 32 * 4);
  float*  qkv_ws  = (float*)alloc((size_t)B_ * N_ * 3 * DIM_ * 4);
  bf16_t* qb      = (bf16_t*)alloc((size_t)B_ * H_ * N_ * D_ * 2);
  bf16_t* kfull   = (bf16_t*)alloc((size_t)B_ * H_ * KF_ * D_ * 2);
  bf16_t* vT      = (bf16_t*)alloc((size_t)B_ * H_ * D_ * KF_ * 2);
  bf16_t* kmid_b  = (bf16_t*)alloc((size_t)B_ * H_ * NM_ * D_ * 2);
  bf16_t* vmidT_b = (bf16_t*)alloc((size_t)B_ * H_ * D_ * NM_ * 2);
  float*  k_m     = (float*)alloc((size_t)B_ * H_ * NM_ * D_ * 4);
  float*  v_m     = (float*)alloc((size_t)B_ * H_ * NM_ * D_ * 4);
  float*  scale_m = (float*)alloc((size_t)B_ * NM_ * H_ * 4);
  float*  idv_out = (float*)alloc((size_t)B_ * NM_ * DIM_ * 4);
  float*  xo      = (float*)alloc((size_t)B_ * N_ * DIM_ * 4);
  bf16_t* xob     = (bf16_t*)alloc((size_t)B_ * N_ * DIM_ * 2);
  float*  route   = (float*)alloc((size_t)B_ * N_ * 4 * 4);
  float*  coef    = (float*)alloc((size_t)B_ * N_ * 32 * 4);

  // prep
  cvt_bf16_kernel<<<g1((size_t)B_ * N_ * DIM_), 256, 0, stream>>>(x, xb, (size_t)B_ * N_ * DIM_);
  cvt_bf16_kernel<<<g1((size_t)B_ * NM_ * DIM_), 256, 0, stream>>>(id_tot, idb, (size_t)B_ * NM_ * DIM_);
  cvt_bf16_kernel<<<g1((size_t)DIM_ * DIM_), 256, 0, stream>>>(W_proj, Wproj_b, (size_t)DIM_ * DIM_);
  fold_w_kernel<<<g1((size_t)3 * DIM_ * DIM_), 256, 0, stream>>>(W_qkv, qkv_A, qkv_B, Wqkv_e, 3 * DIM_, DIM_, DIM_);
  fold_w_kernel<<<g1((size_t)DIM_ * DIM_), 256, 0, stream>>>(W_idv, idv_A, idv_B, Widv_e, DIM_, DIM_, DIM_);
  fold_idk_kernel<<<g1((size_t)H_ * DIM_), 256, 0, stream>>>(W_idk, idk_A, idk_B, Widk_e);
  bcat_kernel<<<g1((size_t)DIM_ * 32), 256, 0, stream>>>(proj_B, Bcat);
  scatter_mem_kernel<<<g1((size_t)B_ * H_ * MEM_ * D_), 256, 0, stream>>>(mem_k, mem_v, kfull, vT);

  // fused-LoRA qkv GEMM: [5120,768] x [2304,768]^T ; each workgroup covers 16x256
  dim3 gb(32, 4, 1);
  gemm_tt_bf16<<<dim3(3 * DIM_ / 256, B_ * N_ / 16), gb, 0, stream>>>(
      xb, Wqkv_e, b_qkv, qkv_ws, B_ * N_, 3 * DIM_, DIM_);
  scatter_qkv_kernel<<<g1((size_t)B_ * N_ * 3 * DIM_), 256, 0, stream>>>(qkv_ws, qb, kfull, vT, k_m, v_m);

  // identity-token projections
  gemm_tt_bf16<<<dim3(DIM_ / 256, B_ * NM_ / 16), gb, 0, stream>>>(
      idb, Widv_e, b_idv, idv_out, B_ * NM_, DIM_, DIM_);
  idk_scale_kernel<<<g1((size_t)B_ * NM_ * H_), 256, 0, stream>>>(id_tot, Widk_e, b_idk, scale_m);
  combine_mid_kernel<<<g1((size_t)B_ * H_ * NM_ * D_), 256, 0, stream>>>(
      k_m, v_m, scale_m, idv_out, out_kmid, out_vmid, kmid_b, vmidT_b, vT);

  // attention
  const int lds_m = 8 * 16 * NM_ * (int)sizeof(float);   // 128 KB
  const int lds_s = 16 * KF_ * (int)sizeof(float);       // 160 KB
  (void)hipFuncSetAttribute((const void*)attn_m_kernel, hipFuncAttributeMaxDynamicSharedMemorySize, lds_m);
  (void)hipFuncSetAttribute((const void*)attn_s_kernel, hipFuncAttributeMaxDynamicSharedMemorySize, lds_s);
  attn_m_kernel<<<dim3(B_ * H_), dim3(32, 8), lds_m, stream>>>(qb, kmid_b, vmidT_b, xo);
  attn_s_kernel<<<dim3(NS_ / 16, H_, B_), dim3(32, 8), lds_s, stream>>>(qb, kfull, vT, xo);

  // routed-LoRA projection
  route_kernel<<<g1((size_t)B_ * N_), 256, 0, stream>>>(xo, route_W, route);
  coef_kernel<<<g1((size_t)B_ * N_ * 32), 256, 0, stream>>>(xo, proj_A, route, coef);
  cvt_bf16_kernel<<<g1((size_t)B_ * N_ * DIM_), 256, 0, stream>>>(xo, xob, (size_t)B_ * N_ * DIM_);
  gemm_tt_bf16<<<dim3(DIM_ / 256, B_ * N_ / 16), gb, 0, stream>>>(
      xob, Wproj_b, b_proj, out, B_ * N_, DIM_, DIM_);
  epilogue_kernel<<<g1((size_t)B_ * N_ * DIM_), 256, 0, stream>>>(coef, Bcat, out);
}
